// LSTM_FEAT_3_24524263260276
// MI455X (gfx1250) — matrix operand order repaired and checked by reading, not compiled, not run on hardware
//
#include <hip/hip_runtime.h>
#include <math.h>

// ---------------------------------------------------------------------------
// LSTM scan, MI455X (gfx1250). Persistent-kernel design:
//   * weights converted f32 -> bf16 once per launch (63 MB, lives in 192MB L2)
//   * all matvecs done with v_wmma_f32_16x16x32_bf16, 16-row tiles
//   * 96 blocks x 128 threads = 384 waves == 384 gate tiles (6144 rows / 16)
//   * grid-wide barrier = monotonic atomic counter in d_ws (reset by init krn)
//   * LSTM cell state c kept in registers (thread i owns unit i every step)
// ---------------------------------------------------------------------------

#define T_STEPS 2048
#define IN_DIM  118
#define XT_DIM  256      // 118 input + 128 embed, zero-padded to 256
#define HID     1536
#define G4      6144     // 4*HID
#define HID2    1024
#define OUT_DIM 128
#define NBLK    96
#define TPB     128      // 4 waves per block -> 384 waves total

typedef __bf16  bf16x16 __attribute__((ext_vector_type(16)));
typedef float   f32x8   __attribute__((ext_vector_type(8)));
typedef unsigned uint4v __attribute__((ext_vector_type(4)));

union ABunion { uint4v u[2]; bf16x16 v; };

// A-tile: 16 rows x 32 K of row-major bf16 W.  ISA 16-bit A layout:
// lanes 0-15: row=lane,   K = kbase+{0..7, 16..23}
// lanes 16-31: row=lane-16, K = kbase+{8..15, 24..31}
__device__ __forceinline__ bf16x16 load_a(const __bf16* W, int ldk,
                                          int rowbase, int kbase, int lane) {
    int row  = rowbase + (lane & 15);
    int koff = kbase + ((lane & 16) ? 8 : 0);
    const __bf16* p = W + (size_t)row * ldk + koff;
    ABunion t;
    t.u[0] = *(const uint4v*)p;          // 8 bf16
    t.u[1] = *(const uint4v*)(p + 16);   // 8 bf16
    return t.v;
}

// B-tile: activation vector replicated across all 16 columns.
// lanes 0-15 hold x[kbase+{0..7,16..23}], lanes 16-31 hold x[kbase+{8..15,24..31}].
__device__ __forceinline__ bf16x16 load_b(const __bf16* x, int kbase, int lane) {
    const __bf16* p = x + kbase + ((lane & 16) ? 8 : 0);
    ABunion t;
    t.u[0] = *(const uint4v*)p;
    t.u[1] = *(const uint4v*)(p + 16);
    return t.v;
}

__device__ __forceinline__ f32x8 wmma_bf16(bf16x16 a, bf16x16 b, f32x8 c) {
    return __builtin_amdgcn_wmma_f32_16x16x32_bf16(
        /*neg_a=*/false, a, /*neg_b=*/false, b,
        /*c_mod=*/(short)0, c, /*reuse_a=*/false, /*reuse_b=*/false);
}

// C/D layout: lanes 0-15 vgpr m -> row rowbase+m ; lanes 16-31 -> rowbase+8+m
__device__ __forceinline__ f32x8 bias2(const float* b1, const float* b2,
                                       int rowbase, int lane) {
    int r0 = rowbase + ((lane & 16) ? 8 : 0);
    f32x8 c;
#pragma unroll
    for (int m = 0; m < 8; ++m) c[m] = b1[r0 + m] + b2[r0 + m];
    return c;
}
__device__ __forceinline__ f32x8 bias1(const float* b1, int rowbase, int lane) {
    int r0 = rowbase + ((lane & 16) ? 8 : 0);
    f32x8 c;
#pragma unroll
    for (int m = 0; m < 8; ++m) c[m] = b1[r0 + m];
    return c;
}

__device__ __forceinline__ float sigmoidf_(float x) { return 1.0f / (1.0f + expf(-x)); }

// Grid-wide barrier: monotonic counter, epoch tracked in registers.
__device__ __forceinline__ void gridbar(unsigned* ctr, unsigned& epoch) {
    __threadfence();
    __syncthreads();
    ++epoch;
    if (threadIdx.x == 0) {
        unsigned target = epoch * (unsigned)NBLK;
        __hip_atomic_fetch_add(ctr, 1u, __ATOMIC_RELEASE, __HIP_MEMORY_SCOPE_AGENT);
        while (__hip_atomic_load(ctr, __ATOMIC_ACQUIRE, __HIP_MEMORY_SCOPE_AGENT) < target) {
            __builtin_amdgcn_s_sleep(1);
        }
    }
    __syncthreads();
    __threadfence();
}

// ------------------------- weight conversion -------------------------------
__global__ void conv_pad(const float* __restrict__ src, __bf16* __restrict__ dst,
                         int rows, int srcK, int dstK) {
    long total = (long)rows * dstK;
    long stride = (long)gridDim.x * blockDim.x;
    for (long idx = (long)blockIdx.x * blockDim.x + threadIdx.x; idx < total; idx += stride) {
        int col  = (int)(idx % dstK);
        long row = idx / dstK;
        dst[idx] = (col < srcK) ? (__bf16)src[row * srcK + col] : (__bf16)0.0f;
    }
}

// ------------------------- per-launch state init ---------------------------
__global__ void init_state(const float* __restrict__ in, const float* __restrict__ h0in,
                           __bf16* xt, __bf16* h0b, __bf16* h1b, unsigned* ctr) {
    int i = blockIdx.x * blockDim.x + threadIdx.x;
    if (i == 0) *ctr = 0u;
    if (i < IN_DIM)       xt[i] = (__bf16)in[i];       // input row 0
    else if (i < XT_DIM)  xt[i] = (__bf16)0.0f;        // embed=0 + K-padding
    if (i < HID) {
        h0b[i] = (__bf16)h0in[i];
        h1b[i] = (__bf16)h0in[HID + i];
    }
}

// ------------------------- persistent scan kernel --------------------------
__global__ void __launch_bounds__(TPB)
lstm_scan(const float* __restrict__ in,
          const float* __restrict__ c0in,
          const float* __restrict__ bih0, const float* __restrict__ bhh0,
          const float* __restrict__ bih1, const float* __restrict__ bhh1,
          const float* __restrict__ bhl,  const float* __restrict__ bout,
          const float* __restrict__ bmap,
          const __bf16* __restrict__ Wih0p, const __bf16* __restrict__ Whh0,
          const __bf16* __restrict__ Wih1,  const __bf16* __restrict__ Whh1,
          const __bf16* __restrict__ Whl,   const __bf16* __restrict__ Wout,
          const __bf16* __restrict__ Wmap,
          __bf16* xt, __bf16* h0b, __bf16* h1b, __bf16* zb,
          float* gates0, float* gates1,
          unsigned* ctr, float* __restrict__ out) {
    const int lane = threadIdx.x & 31;
    const int gw   = (blockIdx.x << 2) + (threadIdx.x >> 5);   // global wave 0..383
    const int gid  = blockIdx.x * TPB + threadIdx.x;

    // cell state lives in registers: thread gid owns hidden unit gid forever
    float c0r = (gid < HID) ? c0in[gid]       : 0.0f;
    float c1r = (gid < HID) ? c0in[HID + gid] : 0.0f;
    unsigned epoch = 0;

    __shared__ __align__(32) float  ys[OUT_DIM];
    __shared__ __align__(32) float  red[OUT_DIM];
    __shared__ __align__(32) __bf16 ysb[OUT_DIM];

    for (int t = 0; t < T_STEPS; ++t) {
        // ---- Stage A: gates0 = Wih0p*xt + Whh0*h0b + b  (384 tiles) ----
        {
            const int rowbase = gw << 4;
            f32x8 acc = bias2(bih0, bhh0, rowbase, lane);
#pragma unroll
            for (int kb = 0; kb < XT_DIM; kb += 32)
                acc = wmma_bf16(load_a(Wih0p, XT_DIM, rowbase, kb, lane),
                                load_b(xt, kb, lane), acc);
            for (int kb = 0; kb < HID; kb += 32)
                acc = wmma_bf16(load_a(Whh0, HID, rowbase, kb, lane),
                                load_b(h0b, kb, lane), acc);
            if (!(lane & 15))
                *(f32x8*)(gates0 + rowbase + ((lane & 16) ? 8 : 0)) = acc;
        }
        gridbar(ctr, epoch);

        // ---- Stage B: LSTM cell 0 (thread-per-unit, c in registers) ----
        if (gid < HID) {
            float iS = sigmoidf_(gates0[gid]);
            float fS = sigmoidf_(gates0[HID + gid]);
            float gS = tanhf(gates0[2 * HID + gid]);
            float oS = sigmoidf_(gates0[3 * HID + gid]);
            c0r = fS * c0r + iS * gS;
            h0b[gid] = (__bf16)(oS * tanhf(c0r));
        }
        gridbar(ctr, epoch);

        // ---- Stage C: gates1 = Wih1*h0' + Whh1*h1 + b ----
        {
            const int rowbase = gw << 4;
            f32x8 acc = bias2(bih1, bhh1, rowbase, lane);
            for (int kb = 0; kb < HID; kb += 32) {
                acc = wmma_bf16(load_a(Wih1, HID, rowbase, kb, lane),
                                load_b(h0b, kb, lane), acc);
                acc = wmma_bf16(load_a(Whh1, HID, rowbase, kb, lane),
                                load_b(h1b, kb, lane), acc);
            }
            if (!(lane & 15))
                *(f32x8*)(gates1 + rowbase + ((lane & 16) ? 8 : 0)) = acc;
        }
        gridbar(ctr, epoch);

        // ---- Stage D: LSTM cell 1 ----
        if (gid < HID) {
            float iS = sigmoidf_(gates1[gid]);
            float fS = sigmoidf_(gates1[HID + gid]);
            float gS = tanhf(gates1[2 * HID + gid]);
            float oS = sigmoidf_(gates1[3 * HID + gid]);
            c1r = fS * c1r + iS * gS;
            h1b[gid] = (__bf16)(oS * tanhf(c1r));
        }
        gridbar(ctr, epoch);

        // ---- Stage E: z = Whl*h1' + b_hl  (64 tiles) ----
        if (gw < (HID2 >> 4)) {
            const int rowbase = gw << 4;
            f32x8 acc = bias1(bhl, rowbase, lane);
            for (int kb = 0; kb < HID; kb += 32)
                acc = wmma_bf16(load_a(Whl, HID, rowbase, kb, lane),
                                load_b(h1b, kb, lane), acc);
            if (!(lane & 15)) {
                __bf16* p = zb + rowbase + ((lane & 16) ? 8 : 0);
#pragma unroll
                for (int m = 0; m < 8; ++m) p[m] = (__bf16)acc[m];
            }
        }
        gridbar(ctr, epoch);

        // ---- Stage F (block 0): y, softmax, output, embed, next xt ----
        if (blockIdx.x == 0) {
            const int w = threadIdx.x >> 5;        // waves 0..3, 2 tiles each
#pragma unroll
            for (int tt = 0; tt < 2; ++tt) {
                const int rowbase = (w + 4 * tt) << 4;
                f32x8 acc = bias1(bout, rowbase, lane);
#pragma unroll
                for (int kb = 0; kb < HID2; kb += 32)
                    acc = wmma_bf16(load_a(Wout, HID2, rowbase, kb, lane),
                                    load_b(zb, kb, lane), acc);
                if (!(lane & 15)) {
                    float* p = ys + rowbase + ((lane & 16) ? 8 : 0);
#pragma unroll
                    for (int m = 0; m < 8; ++m) p[m] = acc[m];
                }
            }
            __syncthreads();

            // softmax over 128 (128 threads, shared tree reductions)
            const int i = threadIdx.x;
            float v = ys[i];
            red[i] = v;
            __syncthreads();
            for (int s = 64; s > 0; s >>= 1) {
                if (i < s) red[i] = fmaxf(red[i], red[i + s]);
                __syncthreads();
            }
            float mx = red[0];
            __syncthreads();
            float e = expf(v - mx);
            red[i] = e;
            __syncthreads();
            for (int s = 64; s > 0; s >>= 1) {
                if (i < s) red[i] += red[i + s];
                __syncthreads();
            }
            float sm = e / red[0];
            out[(size_t)t * OUT_DIM + i] = sm;
            ysb[i] = (__bf16)sm;
            __syncthreads();

            // embed = Wmap*softmax(y) + b_map  -> xt[118..245]
#pragma unroll
            for (int tt = 0; tt < 2; ++tt) {
                const int rowbase = (w + 4 * tt) << 4;
                f32x8 acc = bias1(bmap, rowbase, lane);
#pragma unroll
                for (int kb = 0; kb < OUT_DIM; kb += 32)
                    acc = wmma_bf16(load_a(Wmap, OUT_DIM, rowbase, kb, lane),
                                    load_b(ysb, kb, lane), acc);
                if (!(lane & 15)) {
                    const int r0 = rowbase + ((lane & 16) ? 8 : 0);
#pragma unroll
                    for (int m = 0; m < 8; ++m)
                        xt[IN_DIM + r0 + m] = (__bf16)acc[m];
                }
            }
            // next input row -> xt[0..117]
            if (t + 1 < T_STEPS && threadIdx.x < IN_DIM)
                xt[threadIdx.x] = (__bf16)in[(size_t)(t + 1) * IN_DIM + threadIdx.x];
        }
        gridbar(ctr, epoch);
    }
}

// ---------------------------------------------------------------------------
extern "C" void kernel_launch(void* const* d_in, const int* in_sizes, int n_in,
                              void* d_out, int out_size, void* d_ws, size_t ws_size,
                              hipStream_t stream) {
    (void)in_sizes; (void)n_in; (void)out_size; (void)ws_size;

    const float* in    = (const float*)d_in[0];
    const float* h0in  = (const float*)d_in[1];
    const float* c0in  = (const float*)d_in[2];
    const float* W_ih0 = (const float*)d_in[3];
    const float* W_hh0 = (const float*)d_in[4];
    const float* b_ih0 = (const float*)d_in[5];
    const float* b_hh0 = (const float*)d_in[6];
    const float* W_ih1 = (const float*)d_in[7];
    const float* W_hh1 = (const float*)d_in[8];
    const float* b_ih1 = (const float*)d_in[9];
    const float* b_hh1 = (const float*)d_in[10];
    const float* W_hl  = (const float*)d_in[11];
    const float* b_hl  = (const float*)d_in[12];
    const float* W_out = (const float*)d_in[13];
    const float* b_out = (const float*)d_in[14];
    const float* W_map = (const float*)d_in[15];
    const float* b_map = (const float*)d_in[16];
    float* out = (float*)d_out;

    // workspace carve-up (256-byte aligned chunks); total ~63.3 MB
    char* base = (char*)d_ws;
    size_t o = 0;
    auto take = [&](size_t bytes) -> char* {
        o = (o + 255) & ~(size_t)255;
        char* p = base + o;
        o += bytes;
        return p;
    };
    __bf16* wih0p = (__bf16*)take((size_t)G4 * XT_DIM * 2);
    __bf16* whh0  = (__bf16*)take((size_t)G4 * HID * 2);
    __bf16* wih1  = (__bf16*)take((size_t)G4 * HID * 2);
    __bf16* whh1  = (__bf16*)take((size_t)G4 * HID * 2);
    __bf16* whl   = (__bf16*)take((size_t)HID2 * HID * 2);
    __bf16* wout  = (__bf16*)take((size_t)OUT_DIM * HID2 * 2);
    __bf16* wmap  = (__bf16*)take((size_t)OUT_DIM * OUT_DIM * 2);
    __bf16* xt    = (__bf16*)take(XT_DIM * 2);
    __bf16* h0b   = (__bf16*)take(HID * 2);
    __bf16* h1b   = (__bf16*)take(HID * 2);
    __bf16* zb    = (__bf16*)take(HID2 * 2);
    float*  g0    = (float*) take(G4 * 4);
    float*  g1    = (float*) take(G4 * 4);
    unsigned* ctr = (unsigned*)take(256);

    // one-time (per launch) f32 -> bf16 weight conversion, K-padded for layer0
    conv_pad<<<1024, 256, 0, stream>>>(W_ih0, wih0p, G4, IN_DIM + OUT_DIM, XT_DIM);
    conv_pad<<<2048, 256, 0, stream>>>(W_hh0, whh0, G4, HID, HID);
    conv_pad<<<2048, 256, 0, stream>>>(W_ih1, wih1, G4, HID, HID);
    conv_pad<<<2048, 256, 0, stream>>>(W_hh1, whh1, G4, HID, HID);
    conv_pad<<<1024, 256, 0, stream>>>(W_hl,  whl,  HID2, HID, HID);
    conv_pad<<<128,  256, 0, stream>>>(W_out, wout, OUT_DIM, HID2, HID2);
    conv_pad<<<32,   256, 0, stream>>>(W_map, wmap, OUT_DIM, OUT_DIM, OUT_DIM);

    init_state<<<6, 256, 0, stream>>>(in, h0in, xt, h0b, h1b, ctr);

    lstm_scan<<<NBLK, TPB, 0, stream>>>(
        in, c0in, b_ih0, b_hh0, b_ih1, b_hh1, b_hl, b_out, b_map,
        wih0p, whh0, wih1, whh1, whl, wout, wmap,
        xt, h0b, h1b, zb, g0, g1, ctr, out);
}
